// Model_64304250356258
// MI455X (gfx1250) — matrix operand order repaired and checked by reading, not compiled, not run on hardware
//
#include <hip/hip_runtime.h>
#include <math.h>

#define KW   15.0f
#define EPSR 1e-8f
#define MB   4096   // boundary points staged in LDS (harness uses exactly 4096)

typedef __attribute__((ext_vector_type(16))) _Float16 v16h;
typedef __attribute__((ext_vector_type(8)))  _Float16 v8h;
typedef __attribute__((ext_vector_type(8)))  float    v8f;

// ---------------------------------------------------------------------------
// Kernel 1: pad 50x50 f32 weights (W2,W3,W4) to f16 and shuffle into
// fragment-major WMMA B-layout:
//   Wp[((layer*8 + t*2 + kc)*32 + lane)*16 + e]  =  W[k][n]
// with khi = lane>>4, n = t*16 + (lane&15), k = kc*32 + khi*16 + e,
// contiguous k order shared with the A fragments (lane group khi holds K=16*khi..+15,
// VGPR v holds K = 16*khi + 2v + {0,1}  ==>  half index e = k - 16*khi).
// Each lane in the MLP kernel then loads its fragment as one aligned v16h.
// ---------------------------------------------------------------------------
__global__ void pack_weights_kernel(const float* __restrict__ W2,
                                    const float* __restrict__ W3,
                                    const float* __restrict__ W4,
                                    _Float16* __restrict__ Wp /* 3*8*32*16 */)
{
    int i = blockIdx.x * blockDim.x + threadIdx.x;
    if (i >= 3 * 8 * 32 * 16) return;
    int e    = i & 15;
    int lane = (i >> 4) & 31;
    int frag = i >> 9;              // 0..23 = layer*8 + t*2 + kc
    int l    = frag >> 3;
    int rem  = frag & 7;
    int t    = rem >> 1;
    int kc   = rem & 1;
    int khi  = lane >> 4;
    int n    = t * 16 + (lane & 15);
    int k    = kc * 32 + khi * 16 + e;
    const float* src = (l == 0) ? W2 : ((l == 1) ? W3 : W4);
    float v = (k < 50 && n < 50) ? src[k * 50 + n] : 0.0f;
    Wp[i] = (_Float16)v;
}

// ---------------------------------------------------------------------------
// Kernel 2: boundary-density MLP via WMMA. One wave per 16-row tile.
// act[16x64] f16 in LDS (row-major); hidden layers as 4 N-tiles x 2 K-chunks
// of v_wmma_f32_16x16x32_f16 with f32 accumulation, tanh, write back.
// A-fragments: two aligned ds_load_b128 per K-chunk (row-major act makes the
// per-lane halves contiguous in k). B-fragments: one v16h global load from
// the fragment-major packed weights.
// ---------------------------------------------------------------------------
__global__ void __launch_bounds__(32)
mlp_wmma_kernel(const float* __restrict__ bp,                 // [M][2]
                const float* __restrict__ W1,                 // [2][50]
                const float* __restrict__ b1,                 // [50]
                const _Float16* __restrict__ Wp,              // packed, see above
                const float* __restrict__ b2,
                const float* __restrict__ b3,
                const float* __restrict__ b4,
                const float* __restrict__ W5,                 // [50]
                const float* __restrict__ b5,                 // [1]
                float* __restrict__ hv)                       // [M]
{
    __shared__ _Float16 act[16 * 64];
    const int lane = threadIdx.x;       // 0..31 (wave32)
    const int row0 = blockIdx.x * 16;

    // ---- layer 1 (K=2): scalar, tanh, padded to 64 cols ----
    for (int i = lane; i < 16 * 64; i += 32) {
        int m = i >> 6, c = i & 63;
        float v = 0.0f;
        if (c < 50) {
            float x0 = bp[(row0 + m) * 2 + 0];
            float x1 = bp[(row0 + m) * 2 + 1];
            v = tanhf(fmaf(x1, W1[50 + c], fmaf(x0, W1[c], b1[c])));
        }
        act[i] = (_Float16)v;
    }
    __syncthreads();

    const int mrow = lane & 15;     // A row / B,C,D column index
    const int khi  = lane >> 4;     // lane-group selector per ISA layouts
    const v16h* Wf = (const v16h*)Wp;

    for (int layer = 0; layer < 3; ++layer) {
        const float* bias = (layer == 0) ? b2 : ((layer == 1) ? b3 : b4);

        // A fragments: per lane, K-chunk kc needs halves
        //   act[mrow*64 + kc*32 + khi*16 + 0..7]  and
        //   act[mrow*64 + kc*32 + khi*16 + 8 + 0..7]
        // -> four aligned 16B LDS loads total.
        const v8h* av = (const v8h*)act;
        int base = mrow * 8 + khi * 2;          // in v8h units
        v8h a00 = av[base + 0];
        v8h a01 = av[base + 1];
        v8h a10 = av[base + 4];
        v8h a11 = av[base + 5];
        v16h a0 = __builtin_shufflevector(a00, a01, 0,1,2,3,4,5,6,7,
                                                    8,9,10,11,12,13,14,15);
        v16h a1 = __builtin_shufflevector(a10, a11, 0,1,2,3,4,5,6,7,
                                                    8,9,10,11,12,13,14,15);
        __syncthreads();   // all act reads done before writeback below

        v8f acc[4];
#pragma unroll
        for (int t = 0; t < 4; ++t) {
            int ncol = t * 16 + mrow;
            float bb = (ncol < 50) ? bias[ncol] : 0.0f;   // C layout: value depends on N only
            acc[t] = (v8f){bb, bb, bb, bb, bb, bb, bb, bb};
            v16h bf0 = Wf[(layer * 8 + t * 2 + 0) * 32 + lane];
            v16h bf1 = Wf[(layer * 8 + t * 2 + 1) * 32 + lane];
            acc[t] = __builtin_amdgcn_wmma_f32_16x16x32_f16(
                false, a0, false, bf0, (short)0, acc[t], false, false);
            acc[t] = __builtin_amdgcn_wmma_f32_16x16x32_f16(
                false, a1, false, bf1, (short)0, acc[t], false, false);
        }

        // tanh + writeback per C/D layout: lane-group khi holds M = v + 8*khi, N fixed.
#pragma unroll
        for (int t = 0; t < 4; ++t) {
#pragma unroll
            for (int v = 0; v < 8; ++v) {
                int m = v + khi * 8;
                int n = t * 16 + mrow;
                act[m * 64 + n] = (_Float16)tanhf(acc[t][v]);
            }
        }
        __syncthreads();
    }

    // ---- layer 5: [16x50] . [50] + b5 ----
    if (lane < 16) {
        float s = b5[0];
        for (int c = 0; c < 50; ++c)
            s += (float)act[lane * 64 + c] * W5[c];
        hv[row0 + lane] = s;
    }
}

// ---------------------------------------------------------------------------
// Kernel 3: the O(P*M) boundary integral (dominant cost, VALU-bound).
// LDS-staged boundary data; each thread owns one query point and streams all
// M boundary entries. Branchless both-branch Y1 with select.
// ---------------------------------------------------------------------------
__global__ void __launch_bounds__(256)
boundary_integral_kernel(const float* __restrict__ pts,   // [P][2]
                         const float* __restrict__ bp,    // [M][2]
                         const float* __restrict__ nrm,   // [M][2]
                         const float* __restrict__ dyp,   // scalar
                         const float* __restrict__ hv,    // [M]
                         float* __restrict__ out,         // [P]
                         int P, int M)
{
    __shared__ float sbx[MB];
    __shared__ float sby[MB];
    __shared__ float sh[MB];
    __shared__ unsigned char sc[MB];   // normal code: 0:(0,1) 1:(0,-1) 2:(-1,0) 3:(1,0)

    for (int i = threadIdx.x; i < M; i += blockDim.x) {
        sbx[i] = bp[2 * i + 0];
        sby[i] = bp[2 * i + 1];
        sh[i]  = hv[i];
        float nx = nrm[2 * i + 0], ny = nrm[2 * i + 1];
        sc[i] = (ny > 0.5f) ? 0 : ((ny < -0.5f) ? 1 : ((nx < -0.5f) ? 2 : 3));
    }
    __syncthreads();

    int p = blockIdx.x * blockDim.x + threadIdx.x;
    if (p >= P) return;

    const float px = pts[2 * p + 0];
    const float py = pts[2 * p + 1];
    const float dy = dyp[0];

    float acc = 0.0f;
#pragma unroll 4
    for (int m = 0; m < M; ++m) {
        float dx  = px - sbx[m];
        float dyv = py - sby[m];
        float r   = sqrtf(fmaf(dx, dx, dyv * dyv));
        unsigned char c = sc[m];
        float dot  = (c == 0) ? dyv : ((c == 1) ? -dyv : ((c == 2) ? -dx : dx));
        float drdn = dot / (r + EPSR);

        float xs  = fmaxf(KW * r, 1e-12f);
        float inv = 1.0f / xs;           // reused for 1/x, 8/x, (2/pi)/x
        float y   = xs * xs;

        // --- small branch (x < 8): Y1 rational + (2/pi)*(J1*ln(x) - 1/x) ---
        float num = xs * fmaf(y, fmaf(y, fmaf(y, fmaf(y, fmaf(y, 8.511937935e4f,
                       -4.237922726e7f), 7.349264551e9f), -5.153438139e11f),
                        1.275274390e13f), -4.900604943e13f);
        float den = fmaf(y, fmaf(y, fmaf(y, fmaf(y, fmaf(y, (3.549632885e3f + y),
                       1.020426050e6f), 2.245904002e8f), 3.733650367e10f),
                        4.244419664e12f), 2.499580570e14f);
        float jn = xs * fmaf(y, fmaf(y, fmaf(y, fmaf(y, fmaf(y, -30.16036606f,
                       15704.48260f), -2972611.439f), 242396853.1f),
                       -7895059235.0f), 72362614232.0f);
        float jd = fmaf(y, fmaf(y, fmaf(y, fmaf(y, fmaf(y, 1.0f,
                       376.9991397f), 99447.43394f), 18583304.74f),
                       2300535178.0f), 144725228442.0f);
        float j1s    = jn / jd;
        float ysmall = num / den + 0.636619772f * (j1s * __logf(xs) - inv);

        // --- large branch (x >= 8): asymptotic ---
        float z  = 8.0f * inv;
        float y2 = z * z;
        float xx = xs - 2.356194491f;
        float p1 = fmaf(y2, fmaf(y2, fmaf(y2, fmaf(y2, -0.240337019e-6f,
                       0.2457520174e-5f), -0.3516396496e-4f), 0.183105e-2f), 1.0f);
        float p2 = fmaf(y2, fmaf(y2, fmaf(y2, fmaf(y2, 0.105787412e-6f,
                       -0.88228987e-6f), 0.8449199096e-5f), -0.2002690873e-3f),
                       0.04687499995f);
        float ylarge = sqrtf(0.636619772f * inv) *
                       (__sinf(xx) * p1 + z * __cosf(xx) * p2);

        float y1 = (xs < 8.0f) ? ysmall : ylarge;
        acc = fmaf(y1 * drdn, sh[m], acc);
    }
    out[p] = acc * (0.25f * KW) * dy;
}

// ---------------------------------------------------------------------------
// Host-side launcher.
// Input order: 0 points, 1 boundary_points, 2 normals, 3 dy, 4 W1, 5 b1,
//              6 W2, 7 b2, 8 W3, 9 b3, 10 W4, 11 b4, 12 W5, 13 b5
// Workspace:  [0, 24576)  : 3 x 8 x 32 x 16 f16 fragment-packed weights
//             [24576, ...) : M f32 h_vals
// ---------------------------------------------------------------------------
extern "C" void kernel_launch(void* const* d_in, const int* in_sizes, int n_in,
                              void* d_out, int out_size, void* d_ws, size_t ws_size,
                              hipStream_t stream) {
    const float* pts  = (const float*)d_in[0];
    const float* bp   = (const float*)d_in[1];
    const float* nrm  = (const float*)d_in[2];
    const float* dyp  = (const float*)d_in[3];
    const float* W1   = (const float*)d_in[4];
    const float* b1   = (const float*)d_in[5];
    const float* W2   = (const float*)d_in[6];
    const float* b2   = (const float*)d_in[7];
    const float* W3   = (const float*)d_in[8];
    const float* b3   = (const float*)d_in[9];
    const float* W4   = (const float*)d_in[10];
    const float* b4   = (const float*)d_in[11];
    const float* W5   = (const float*)d_in[12];
    const float* b5   = (const float*)d_in[13];
    float* out = (float*)d_out;

    const int P = in_sizes[0] / 2;
    const int M = in_sizes[1] / 2;

    _Float16* Wp = (_Float16*)d_ws;
    float*    hv = (float*)((char*)d_ws + 3 * 8 * 32 * 16 * sizeof(_Float16));

    // 1) pack weights into fragment-major f16 layout
    pack_weights_kernel<<<(3 * 8 * 32 * 16 + 255) / 256, 256, 0, stream>>>(W2, W3, W4, Wp);

    // 2) MLP (WMMA), one wave per 16 boundary rows
    mlp_wmma_kernel<<<M / 16, 32, 0, stream>>>(bp, W1, b1, Wp, b2, b3, b4, W5, b5, hv);

    // 3) boundary integral
    boundary_integral_kernel<<<(P + 255) / 256, 256, 0, stream>>>(
        pts, bp, nrm, dyp, hv, out, P, M);
}
